// Attention_46858093199424
// MI455X (gfx1250) — matrix operand order repaired and checked by reading, not compiled, not run on hardware
//
#include <hip/hip_runtime.h>
#include <hip/hip_bf16.h>

typedef _Float16 f16;
typedef __attribute__((ext_vector_type(16))) _Float16 v16h;
typedef __attribute__((ext_vector_type(8)))  _Float16 v8h;
typedef __attribute__((ext_vector_type(8)))  float    v8f;

#define BB    2
#define NN    2048
#define DIMM  1024
#define HH    16
#define DD    64
#define MMEM  16
#define JPAD  2080          /* 16 mem + 2048 seq + 16 zero pad (multiple of 32) */
#define ROWS  (BB*NN)       /* 4096 */
#define QKS   10.0f

union AFrag { v16h v; v8h h[2]; };

/* ---------------- elementwise converts ---------------- */

__global__ void k_f32_to_f16(const float* __restrict__ src, f16* __restrict__ dst, int n) {
  int i = blockIdx.x * 256 + threadIdx.x;
  if (i < n) dst[i] = (f16)src[i];
}

/* W [K, Nc] row-major f32 -> WT [Nc, K] row-major f16 */
__global__ void k_transpose_to_f16(const float* __restrict__ W, f16* __restrict__ WT,
                                   int K, int Nc) {
  int i = blockIdx.x * 256 + threadIdx.x;
  if (i >= K * Nc) return;
  int k = i / Nc, n = i % Nc;
  WT[(size_t)n * K + k] = (f16)W[i];
}

/* ---------------- WMMA GEMM: C[Mr,Nc] = A[Mr,K] * B, B given transposed as BT[Nc,K] ---- */

__global__ __launch_bounds__(256) void k_gemm_f16(
    const f16* __restrict__ A, const f16* __restrict__ BT, float* __restrict__ C,
    int Mr, int Nc, int K) {
  int tid  = threadIdx.x;
  int wid  = blockIdx.x * 8 + (tid >> 5);
  int lane = tid & 31, lm = lane & 15, hi = lane >> 4;
  int tilesN = Nc >> 6;
  int mt = wid / tilesN, nt = wid % tilesN;
  if (mt * 16 >= Mr) return;

  const f16* arow = A + (size_t)(mt * 16 + lm) * K;
  v8f acc0 = {0,0,0,0,0,0,0,0}, acc1 = acc0, acc2 = acc0, acc3 = acc0;

  for (int kk = 0; kk < K; kk += 32) {
    AFrag a;
    a.h[0] = *(const v8h*)(arow + kk + hi * 16);
    a.h[1] = *(const v8h*)(arow + kk + hi * 16 + 8);
#pragma unroll
    for (int s = 0; s < 4; ++s) {
      const f16* bcol = BT + (size_t)(nt * 64 + s * 16 + lm) * K + kk + hi * 16;
      AFrag b;
      b.h[0] = *(const v8h*)(bcol);
      b.h[1] = *(const v8h*)(bcol + 8);
      v8f& acc = (s == 0 ? acc0 : s == 1 ? acc1 : s == 2 ? acc2 : acc3);
      acc = __builtin_amdgcn_wmma_f32_16x16x32_f16(false, a.v, false, b.v,
                                                   (short)0, acc, false, false);
    }
  }
  float* crow = C + (size_t)(mt * 16 + hi * 8) * Nc + nt * 64 + lm;
#pragma unroll
  for (int r = 0; r < 8; ++r) {
    crow[(size_t)r * Nc +  0] = acc0[r];
    crow[(size_t)r * Nc + 16] = acc1[r];
    crow[(size_t)r * Nc + 32] = acc2[r];
    crow[(size_t)r * Nc + 48] = acc3[r];
  }
}

/* ---------------- l2norm + RoPE, repack q/k/v ----------------
   qf/kf/vf: [ROWS, H*D] f32 (GEMM outputs)
   qh: [B,H,N,D] f16 ; kh: [B,H,JPAD,D] f16 (seq at j=M+i) ; vT: [B,H,D,JPAD] f16 */
__global__ __launch_bounds__(256) void k_prep(
    const float* __restrict__ qf, const float* __restrict__ kf,
    const float* __restrict__ vf, const float* __restrict__ freqs,
    f16* __restrict__ qh, f16* __restrict__ kh, f16* __restrict__ vT) {
  int tid = threadIdx.x;
  int w   = blockIdx.x * 8 + (tid >> 5);   /* 0..65535 = (row, h) */
  int lane = tid & 31;
  int row = w >> 4, h = w & 15;
  int b = row >> 11, i = row & 2047;
  int d = lane * 2;

  size_t off = (size_t)row * DIMM + h * DD + d;
  float q0 = qf[off], q1 = qf[off + 1];
  float k0 = kf[off], k1 = kf[off + 1];
  float v0 = vf[off], v1 = vf[off + 1];

  float sq = q0 * q0 + q1 * q1;
  float sk = k0 * k0 + k1 * k1;
#pragma unroll
  for (int m = 1; m < 32; m <<= 1) { sq += __shfl_xor(sq, m, 32); sk += __shfl_xor(sk, m, 32); }
  float qn = fmaxf(sqrtf(sq), 1e-12f), kn = fmaxf(sqrtf(sk), 1e-12f);
  q0 /= qn; q1 /= qn; k0 /= kn; k1 /= kn;

  float f0 = freqs[(size_t)i * DD + d], f1 = freqs[(size_t)i * DD + d + 1];
  float c0 = __cosf(f0), s0 = __sinf(f0), c1 = __cosf(f1), s1 = __sinf(f1);

  /* rot[d] = (d<32) ? -t[d+32] : t[d-32]  -> partner lane = lane^16 */
  float pq0 = __shfl_xor(q0, 16, 32), pq1 = __shfl_xor(q1, 16, 32);
  float pk0 = __shfl_xor(k0, 16, 32), pk1 = __shfl_xor(k1, 16, 32);
  float pv0 = __shfl_xor(v0, 16, 32), pv1 = __shfl_xor(v1, 16, 32);
  float sgn = (lane < 16) ? -1.0f : 1.0f;

  float oq0 = q0 * c0 + sgn * pq0 * s0, oq1 = q1 * c1 + sgn * pq1 * s1;
  float ok0 = k0 * c0 + sgn * pk0 * s0, ok1 = k1 * c1 + sgn * pk1 * s1;
  float ov0 = v0 * c0 + sgn * pv0 * s0, ov1 = v1 * c1 + sgn * pv1 * s1;

  size_t qoff = ((size_t)(b * HH + h) * NN + i) * DD + d;
  qh[qoff] = (f16)oq0; qh[qoff + 1] = (f16)oq1;
  size_t koff = ((size_t)(b * HH + h) * JPAD + (MMEM + i)) * DD + d;
  kh[koff] = (f16)ok0; kh[koff + 1] = (f16)ok1;
  size_t voff = ((size_t)(b * HH + h) * DD + d) * JPAD + (MMEM + i);
  vT[voff] = (f16)ov0; vT[voff + JPAD] = (f16)ov1;
}

/* memory slots (j<16, l2norm on k, no rope) + zero pad (j in [2064,2080)) */
__global__ void k_mem(const float* __restrict__ mem_k, const float* __restrict__ mem_v,
                      f16* __restrict__ kh, f16* __restrict__ vT) {
  int idx = blockIdx.x * 256 + threadIdx.x;     /* 2*16*32*64 = 65536 */
  int d = idx & 63, slot = (idx >> 6) & 31, h = (idx >> 11) & 15, b = idx >> 15;
  float kval = 0.f, vval = 0.f;
  int j;
  if (slot < 16) {
    j = slot;
    const float* kr = mem_k + ((size_t)h * MMEM + slot) * DD;
    float s = 0.f;
    for (int dd = 0; dd < DD; ++dd) { float t = kr[dd]; s += t * t; }
    kval = kr[d] / fmaxf(sqrtf(s), 1e-12f);
    vval = mem_v[((size_t)h * MMEM + slot) * DD + d];
  } else {
    j = 2048 + slot;                            /* 2064..2079 : zeros */
  }
  kh[((size_t)(b * HH + h) * JPAD + j) * DD + d] = (f16)kval;
  vT[((size_t)(b * HH + h) * DD + d) * JPAD + j] = (f16)vval;
}

/* head-gate GEMV: hg[row,h] = x[row,:] @ Whg[:,h] */
__global__ __launch_bounds__(128) void k_headgate(const float* __restrict__ x,
                                                  const float* __restrict__ Whg,
                                                  float* __restrict__ hg) {
  int w = blockIdx.x * 4 + (threadIdx.x >> 5);
  int lane = threadIdx.x & 31;
  float acc[16];
#pragma unroll
  for (int h = 0; h < 16; ++h) acc[h] = 0.f;
  const float* xr = x + (size_t)w * DIMM;
  for (int k = lane; k < DIMM; k += 32) {
    float xv = xr[k];
    const float* wr = Whg + (size_t)k * HH;
#pragma unroll
    for (int h = 0; h < 16; ++h) acc[h] += xv * wr[h];
  }
#pragma unroll
  for (int h = 0; h < 16; ++h)
    for (int m = 1; m < 32; m <<= 1) acc[h] += __shfl_xor(acc[h], m, 32);
  if (lane == 0) {
#pragma unroll
    for (int h = 0; h < 16; ++h) hg[(size_t)w * HH + h] = acc[h];
  }
}

/* ---------------- fused talking-heads attention ----------------
   one block = 16 waves = 16 heads; one 16-row i-tile; stream j in steps of 32.
   two-pass online softmax of W_pre-mixed logits; W_post mix before attn@v. */
__global__ __launch_bounds__(512) void k_attn(
    const f16* __restrict__ qh, const f16* __restrict__ kh, const f16* __restrict__ vT,
    const float* __restrict__ W_pre, const float* __restrict__ W_post,
    float* __restrict__ attn_out) {
  __shared__ float dotsS[HH][32][16];                 /* [h][j][i]  32 KB */
  __shared__ __align__(16) f16 attn2S[HH][16][32];    /* [g'][i][j] 16 KB */

  int tid = threadIdx.x;
  int h = tid >> 5;                   /* wave = head */
  int lane = tid & 31, lm = lane & 15, hi = lane >> 4;
  int blk = blockIdx.x;
  int b = blk >> 7, it = blk & 127;
  int i0 = it * 16;

  /* q fragments for d=0..31 and d=32..63 held in registers for whole kernel */
  const f16* qrow = qh + ((size_t)(b * HH + h) * NN + i0 + lm) * DD;
  AFrag aq0, aq1;
  aq0.h[0] = *(const v8h*)(qrow + hi * 16);
  aq0.h[1] = *(const v8h*)(qrow + hi * 16 + 8);
  aq1.h[0] = *(const v8h*)(qrow + 32 + hi * 16);
  aq1.h[1] = *(const v8h*)(qrow + 32 + hi * 16 + 8);

  float wpre[16], wpost[16];
#pragma unroll
  for (int g = 0; g < 16; ++g) { wpre[g] = W_pre[h * 16 + g]; wpost[g] = W_post[h * 16 + g]; }

  const f16* kbase = kh + (size_t)(b * HH + h) * JPAD * DD;
  const f16* vbase = vT + (size_t)(b * HH + h) * DD * JPAD;

  const int NJT = JPAD / 32;          /* 65 */
  float m_run = -3.0e38f, l_run = 0.f;

  /* ---- pass 1: softmax statistics ---- */
  for (int jt = 0; jt < NJT; ++jt) {
    int j0 = jt * 32;
#pragma unroll
    for (int sub = 0; sub < 2; ++sub) {
      v8f acc = {0,0,0,0,0,0,0,0};
      const f16* krow = kbase + (size_t)(j0 + sub * 16 + lm) * DD + hi * 16;
      AFrag bk;
      bk.h[0] = *(const v8h*)(krow);
      bk.h[1] = *(const v8h*)(krow + 8);
      acc = __builtin_amdgcn_wmma_f32_16x16x32_f16(false, aq0.v, false, bk.v,
                                                   (short)0, acc, false, false);
      bk.h[0] = *(const v8h*)(krow + 32);
      bk.h[1] = *(const v8h*)(krow + 40);
      acc = __builtin_amdgcn_wmma_f32_16x16x32_f16(false, aq1.v, false, bk.v,
                                                   (short)0, acc, false, false);
#pragma unroll
      for (int r = 0; r < 8; ++r) dotsS[h][sub * 16 + lm][hi * 8 + r] = acc[r];
    }
    __syncthreads();
    /* mix W_pre over heads; lane owns (i=lm, j=hi*16+t) */
    float mloc = -3.0e38f, sv[16];
#pragma unroll
    for (int t = 0; t < 16; ++t) {
      int j = hi * 16 + t;
      float s = 0.f;
#pragma unroll
      for (int g = 0; g < 16; ++g) s += wpre[g] * dotsS[g][j][lm];
      s *= QKS;
      if (j0 + j > i0 + lm + MMEM) s = -3.0e38f;
      sv[t] = s;
      mloc = fmaxf(mloc, s);
    }
    mloc = fmaxf(mloc, __shfl_xor(mloc, 16, 32));
    float m_new = fmaxf(m_run, mloc);
    float sum = 0.f;
#pragma unroll
    for (int t = 0; t < 16; ++t) sum += __expf(sv[t] - m_new);
    sum += __shfl_xor(sum, 16, 32);
    l_run = l_run * __expf(m_run - m_new) + sum;
    m_run = m_new;
    __syncthreads();
  }

  float rcpl = 1.0f / l_run;
  v8f oa0 = {0,0,0,0,0,0,0,0}, oa1 = oa0, oa2 = oa0, oa3 = oa0;

  /* ---- pass 2: normalized attn, W_post mix, attn @ v ---- */
  for (int jt = 0; jt < NJT; ++jt) {
    int j0 = jt * 32;
#pragma unroll
    for (int sub = 0; sub < 2; ++sub) {
      v8f acc = {0,0,0,0,0,0,0,0};
      const f16* krow = kbase + (size_t)(j0 + sub * 16 + lm) * DD + hi * 16;
      AFrag bk;
      bk.h[0] = *(const v8h*)(krow);
      bk.h[1] = *(const v8h*)(krow + 8);
      acc = __builtin_amdgcn_wmma_f32_16x16x32_f16(false, aq0.v, false, bk.v,
                                                   (short)0, acc, false, false);
      bk.h[0] = *(const v8h*)(krow + 32);
      bk.h[1] = *(const v8h*)(krow + 40);
      acc = __builtin_amdgcn_wmma_f32_16x16x32_f16(false, aq1.v, false, bk.v,
                                                   (short)0, acc, false, false);
#pragma unroll
      for (int r = 0; r < 8; ++r) dotsS[h][sub * 16 + lm][hi * 8 + r] = acc[r];
    }
    __syncthreads();
    float av[16];
#pragma unroll
    for (int t = 0; t < 16; ++t) {
      int j = hi * 16 + t;
      float s = 0.f;
#pragma unroll
      for (int g = 0; g < 16; ++g) s += wpre[g] * dotsS[g][j][lm];
      s *= QKS;
      if (j0 + j > i0 + lm + MMEM) s = -3.0e38f;
      av[t] = __expf(s - m_run) * rcpl;
    }
    __syncthreads();                      /* all reads of raw dots done */
#pragma unroll
    for (int t = 0; t < 16; ++t) dotsS[h][hi * 16 + t][lm] = av[t];  /* attn of head h */
    __syncthreads();
#pragma unroll
    for (int t = 0; t < 16; ++t) {
      int j = hi * 16 + t;
      float s = 0.f;
#pragma unroll
      for (int g = 0; g < 16; ++g) s += wpost[g] * dotsS[g][j][lm];
      attn2S[h][lm][j] = (f16)s;
    }
    __syncthreads();
    AFrag aa;
    aa.h[0] = *(const v8h*)&attn2S[h][lm][hi * 16];
    aa.h[1] = *(const v8h*)&attn2S[h][lm][hi * 16 + 8];
#pragma unroll
    for (int s4 = 0; s4 < 4; ++s4) {
      const f16* vrow = vbase + (size_t)(s4 * 16 + lm) * JPAD + j0 + hi * 16;
      AFrag bv;
      bv.h[0] = *(const v8h*)(vrow);
      bv.h[1] = *(const v8h*)(vrow + 8);
      v8f& oacc = (s4 == 0 ? oa0 : s4 == 1 ? oa1 : s4 == 2 ? oa2 : oa3);
      oacc = __builtin_amdgcn_wmma_f32_16x16x32_f16(false, aa.v, false, bv.v,
                                                    (short)0, oacc, false, false);
    }
    __syncthreads();
  }

  float* orow = attn_out + ((size_t)(b * NN) + i0 + hi * 8) * DIMM + h * DD + lm;
#pragma unroll
  for (int r = 0; r < 8; ++r) {
    orow[(size_t)r * DIMM +  0] = oa0[r];
    orow[(size_t)r * DIMM + 16] = oa1[r];
    orow[(size_t)r * DIMM + 32] = oa2[r];
    orow[(size_t)r * DIMM + 48] = oa3[r];
  }
}

/* gates + f16 pack before output projection */
__global__ void k_combine(const float* __restrict__ ao, const float* __restrict__ hg,
                          const float* __restrict__ bhg, const float* __restrict__ gf,
                          const float* __restrict__ bg, f16* __restrict__ oh) {
  int idx = blockIdx.x * 256 + threadIdx.x;
  if (idx >= ROWS * DIMM) return;
  int row = idx >> 10, c = idx & 1023, h = c >> 6;
  float s1 = 1.f / (1.f + __expf(-(hg[(size_t)row * HH + h] + bhg[h])));
  float s2 = 1.f / (1.f + __expf(-(gf[idx] + bg[c])));
  oh[idx] = (f16)(ao[idx] * s1 * s2);
}

/* ---------------- host-side launch ---------------- */

static inline char* wspart(void* ws, size_t& off, size_t bytes) {
  char* p = (char*)ws + off;
  off += (bytes + 255) & ~(size_t)255;
  return p;
}

extern "C" void kernel_launch(void* const* d_in, const int* in_sizes, int n_in,
                              void* d_out, int out_size, void* d_ws, size_t ws_size,
                              hipStream_t stream) {
  (void)in_sizes; (void)n_in; (void)out_size; (void)ws_size;
  const float* x      = (const float*)d_in[0];
  const float* freqs  = (const float*)d_in[1];
  const float* Wq     = (const float*)d_in[2];
  const float* Wk     = (const float*)d_in[3];
  const float* Wv     = (const float*)d_in[4];
  const float* mem_k  = (const float*)d_in[5];
  const float* mem_v  = (const float*)d_in[6];
  const float* W_pre  = (const float*)d_in[7];
  const float* W_post = (const float*)d_in[8];
  const float* Wg     = (const float*)d_in[9];
  const float* bg     = (const float*)d_in[10];
  const float* Whg    = (const float*)d_in[11];
  const float* bhg    = (const float*)d_in[12];
  const float* Wo     = (const float*)d_in[13];

  size_t off = 0;
  f16*   xh   = (f16*)  wspart(d_ws, off, (size_t)ROWS * DIMM * 2);
  f16*   WqT  = (f16*)  wspart(d_ws, off, (size_t)DIMM * DIMM * 2);
  f16*   WkT  = (f16*)  wspart(d_ws, off, (size_t)DIMM * DIMM * 2);
  f16*   WvT  = (f16*)  wspart(d_ws, off, (size_t)DIMM * DIMM * 2);
  f16*   WgT  = (f16*)  wspart(d_ws, off, (size_t)DIMM * DIMM * 2);
  f16*   WoT  = (f16*)  wspart(d_ws, off, (size_t)DIMM * DIMM * 2);
  float* qf   = (float*)wspart(d_ws, off, (size_t)ROWS * DIMM * 4);
  float* kf   = (float*)wspart(d_ws, off, (size_t)ROWS * DIMM * 4);
  float* vf   = (float*)wspart(d_ws, off, (size_t)ROWS * DIMM * 4);
  float* gf   = (float*)wspart(d_ws, off, (size_t)ROWS * DIMM * 4);
  float* hg   = (float*)wspart(d_ws, off, (size_t)ROWS * HH * 4);
  f16*   qhd  = (f16*)  wspart(d_ws, off, (size_t)BB * HH * NN * DD * 2);
  f16*   khd  = (f16*)  wspart(d_ws, off, (size_t)BB * HH * JPAD * DD * 2);
  f16*   vTd  = (f16*)  wspart(d_ws, off, (size_t)BB * HH * DD * JPAD * 2);
  float* ao   = (float*)wspart(d_ws, off, (size_t)ROWS * DIMM * 4);
  f16*   oh   = (f16*)  wspart(d_ws, off, (size_t)ROWS * DIMM * 2);

  const int nElem = ROWS * DIMM;

  k_f32_to_f16<<<(nElem + 255) / 256, 256, 0, stream>>>(x, xh, nElem);
  k_transpose_to_f16<<<(DIMM * DIMM + 255) / 256, 256, 0, stream>>>(Wq, WqT, DIMM, DIMM);
  k_transpose_to_f16<<<(DIMM * DIMM + 255) / 256, 256, 0, stream>>>(Wk, WkT, DIMM, DIMM);
  k_transpose_to_f16<<<(DIMM * DIMM + 255) / 256, 256, 0, stream>>>(Wv, WvT, DIMM, DIMM);
  k_transpose_to_f16<<<(DIMM * DIMM + 255) / 256, 256, 0, stream>>>(Wg, WgT, DIMM, DIMM);
  k_transpose_to_f16<<<(DIMM * DIMM + 255) / 256, 256, 0, stream>>>(Wo, WoT, DIMM, DIMM);

  /* 4096 waves of 16x64 tiles = 512 blocks x 8 waves */
  k_gemm_f16<<<512, 256, 0, stream>>>(xh, WqT, qf, ROWS, DIMM, DIMM);
  k_gemm_f16<<<512, 256, 0, stream>>>(xh, WkT, kf, ROWS, DIMM, DIMM);
  k_gemm_f16<<<512, 256, 0, stream>>>(xh, WvT, vf, ROWS, DIMM, DIMM);
  k_gemm_f16<<<512, 256, 0, stream>>>(xh, WgT, gf, ROWS, DIMM, DIMM);

  k_headgate<<<ROWS / 4, 128, 0, stream>>>(x, Whg, hg);
  k_prep<<<8192, 256, 0, stream>>>(qf, kf, vf, freqs, qhd, khd, vTd);
  k_mem<<<256, 256, 0, stream>>>(mem_k, mem_v, khd, vTd);

  k_attn<<<BB * (NN / 16), 512, 0, stream>>>(qhd, khd, vTd, W_pre, W_post, ao);

  k_combine<<<(nElem + 255) / 256, 256, 0, stream>>>(ao, hg, bhg, gf, bg, oh);
  k_gemm_f16<<<512, 256, 0, stream>>>(oh, WoT, (float*)d_out, ROWS, DIMM, DIMM);
}
